// GeDeformRoIPoolPack_13469017440356
// MI455X (gfx1250) — matrix operand order repaired and checked by reading, not compiled, not run on hardware
//
#include <hip/hip_runtime.h>

#define OH 7
#define OW 7
#define CCH 256
#define SCALE 0.0625f
#define GAMMA 0.1f
#define SIGMA 0.5f
#define FDIM (OH * OW * CCH)   // 12544
#define DFC 1024
#define NB 4
#define NH 100
#define NW 152
#define NROIS 512
#define RESC 98                // OH*OW*2

typedef __attribute__((ext_vector_type(16))) __bf16 v16bf;
typedef __attribute__((ext_vector_type(8)))  __bf16 v8bf;
typedef __attribute__((ext_vector_type(8)))  float  v8f;

// ---------------------------------------------------------------------------
// NCHW -> NHWC transpose so ROI-pool channel gathers are coalesced.
// ---------------------------------------------------------------------------
__global__ void nchw_to_nhwc_kernel(const float* __restrict__ in,
                                    float* __restrict__ out) {
  const long total = (long)NB * CCH * NH * NW;
  for (long i = blockIdx.x * (long)blockDim.x + threadIdx.x; i < total;
       i += (long)gridDim.x * blockDim.x) {
    int c = (int)(i % CCH); long t = i / CCH;
    int x = (int)(t % NW);  t /= NW;
    int y = (int)(t % NH);  int b = (int)(t / NH);
    out[i] = in[(((long)b * CCH + c) * NH + y) * NW + x];
  }
}

// ---------------------------------------------------------------------------
// fp32 (rows x K) -> row-interleaved split-bf16: row r of the destination is
// 2K wide: [ hi(0..K-1) | lo(0..K-1) ].  lo is the bf16 of (x - f32(hi)).
// Templated on K so r = i/K, c = i%K compile to mul-shift.
// ---------------------------------------------------------------------------
template <int K>
__global__ void split_bf16_rows_kernel(const float* __restrict__ src,
                                       __bf16* __restrict__ dst, long n8) {
  for (long g = blockIdx.x * (long)blockDim.x + threadIdx.x; g < n8;
       g += (long)gridDim.x * blockDim.x) {
    long i = g * 8;                 // K % 8 == 0 -> group never spans rows
    long r = i / K;
    int  c = (int)(i - r * K);
    float4 u0 = ((const float4*)(src + i))[0];
    float4 u1 = ((const float4*)(src + i))[1];
    float f[8] = {u0.x, u0.y, u0.z, u0.w, u1.x, u1.y, u1.z, u1.w};
    v8bf h, l;
#pragma unroll
    for (int j = 0; j < 8; ++j) {
      __bf16 hv = (__bf16)f[j];
      h[j] = hv;
      l[j] = (__bf16)(f[j] - (float)hv);
    }
    __bf16* row = dst + r * (2L * K) + c;
    *(v8bf*)(row)     = h;
    *(v8bf*)(row + K) = l;
  }
}

// ---------------------------------------------------------------------------
// Deformable ROI pool. One block per (roi, ph, pw) bin, one thread per channel.
// ---------------------------------------------------------------------------
__global__ void deform_roi_pool_kernel(const float* __restrict__ feat,   // (B,H,W,C)
                                       const float* __restrict__ rois,   // (N,5)
                                       const float* __restrict__ offset, // (N,2,7,7) or null
                                       float* __restrict__ out) {        // (N,C,7,7)
  int bin = blockIdx.x;
  int pw = bin % OW; int t = bin / OW;
  int ph = t % OH;   int n = t / OH;
  int c  = threadIdx.x;

  const float* r = rois + n * 5;
  int   b  = (int)r[0];
  float x1 = r[1] * SCALE - 0.5f;
  float y1 = r[2] * SCALE - 0.5f;
  float x2 = r[3] * SCALE - 0.5f;
  float y2 = r[4] * SCALE - 0.5f;
  float rw = x2 - x1, rh = y2 - y1;
  float bw = rw / OW, bh = rh / OH;
  float sw = x1, sh = y1;
  if (offset) {
    sw += GAMMA * rw * offset[((n * 2 + 0) * OH + ph) * OW + pw];
    sh += GAMMA * rh * offset[((n * 2 + 1) * OH + ph) * OW + pw];
  }

  float acc = 0.0f;
#pragma unroll
  for (int iy = 0; iy < 2; ++iy) {
    float gy = (iy + 0.5f) * 0.5f;
    float y  = sh + ((float)ph + gy) * bh;
#pragma unroll
    for (int ix = 0; ix < 2; ++ix) {
      float gx = (ix + 0.5f) * 0.5f;
      float x  = sw + ((float)pw + gx) * bw;
      bool valid = (y > -1.0f) && (y < (float)NH) && (x > -1.0f) && (x < (float)NW);
      float yc = fminf(fmaxf(y, 0.0f), (float)NH - 1.0f);
      float xc = fminf(fmaxf(x, 0.0f), (float)NW - 1.0f);
      int y0 = min(max((int)floorf(yc), 0), NH - 2);
      int x0 = min(max((int)floorf(xc), 0), NW - 2);
      float ly = yc - (float)y0, lx = xc - (float)x0;
      float hy = 1.0f - ly,      hx = 1.0f - lx;
      const float* base = feat + (((long)b * NH + y0) * NW + x0) * CCH + c;
      float v00 = base[0];
      float v01 = base[CCH];
      float v10 = base[(long)NW * CCH];
      float v11 = base[(long)NW * CCH + CCH];
      float v = hy * hx * v00 + hy * lx * v01 + ly * hx * v10 + ly * lx * v11;
      acc += valid ? v : 0.0f;
    }
  }
  out[(((long)n * CCH + c) * OH + ph) * OW + pw] = acc * 0.25f;
}

// ---------------------------------------------------------------------------
// Only rescale[:,0:4] is ever used -> 4 dot products per roi + roi update.
// ---------------------------------------------------------------------------
__global__ void rescale_rois_kernel(const float* __restrict__ flat,  // (N,FDIM)
                                    const float* __restrict__ w,     // (98,FDIM)
                                    const float* __restrict__ bvec,  // (98)
                                    const float* __restrict__ rois,  // (N,5)
                                    float* __restrict__ new_rois) {  // (N,5)
  __shared__ float red[4][256];
  int n = blockIdx.x;
  int t = threadIdx.x;
  const float* f = flat + (long)n * FDIM;
  float s0 = 0, s1 = 0, s2 = 0, s3 = 0;
  for (int k = t; k < FDIM; k += 256) {
    float fv = f[k];
    s0 += fv * w[0 * FDIM + k];
    s1 += fv * w[1 * FDIM + k];
    s2 += fv * w[2 * FDIM + k];
    s3 += fv * w[3 * FDIM + k];
  }
  red[0][t] = s0; red[1][t] = s1; red[2][t] = s2; red[3][t] = s3;
  __syncthreads();
  for (int off = 128; off > 0; off >>= 1) {
    if (t < off) {
      red[0][t] += red[0][t + off];
      red[1][t] += red[1][t + off];
      red[2][t] += red[2][t + off];
      red[3][t] += red[3][t + off];
    }
    __syncthreads();
  }
  if (t == 0) {
    float r0 = 1.0f + SIGMA / (1.0f + __expf(-(red[0][0] + bvec[0])));
    float r1 = 1.0f + SIGMA / (1.0f + __expf(-(red[1][0] + bvec[1])));
    float r2 = 1.0f + SIGMA / (1.0f + __expf(-(red[2][0] + bvec[2])));
    float r3 = 1.0f + SIGMA / (1.0f + __expf(-(red[3][0] + bvec[3])));
    const float* r = rois + n * 5;
    float cx = (r[1] + r[3]) * 0.5f + r0;
    float cy = (r[2] + r[4]) * 0.5f + r1;
    float nw = (r[3] - r[1]) * r2;
    float nh = (r[4] - r[2]) * r3;
    new_rois[n * 5 + 0] = r[0];
    new_rois[n * 5 + 1] = cx - 0.5f * nw;
    new_rois[n * 5 + 2] = cy - 0.5f * nh;
    new_rois[n * 5 + 3] = cx + 0.5f * nw;
    new_rois[n * 5 + 4] = cy + 0.5f * nh;
  }
}

// ---------------------------------------------------------------------------
// out[m,n] = act( sum_k A[m,k]*W[n,k] + bias[n] ), operands pre-split into
// row-interleaved bf16 hi/lo planes (row stride 2K).  bf16x3 refinement:
//   acc = Ah*Wh + Ah*Wl + Al*Wh  (f32 accumulate) ~ fp32 accuracy.
// One wave owns a 16 x (16*JT) strip.  K is a template constant so every load
// in the hot loop is  base-pointer + literal IOFFSET:
//   A lane fragment:  abase(+0,+8)  hi | abase(+K,+K+8)  lo
//   B tile j:         wbase + j*16*2K (hi) | + K (lo)      (j*64KB..2.4MB,
//                     inside the signed 24-bit instruction offset)
// Loop-carried state: 2 pointers, 16*JT acc VGPRs -> no spills, no per-lane
// 64-bit address VALU in the loop.
// ---------------------------------------------------------------------------
template <int ACT, int JT, int K>
__global__ void gemm_nt_bf16x3_kernel(const __bf16* __restrict__ Ai,  // M x 2K
                                      const __bf16* __restrict__ Wi,  // N x 2K
                                      const float* __restrict__ bias,
                                      float* __restrict__ out,
                                      int M, int N) {
  const int wave = threadIdx.x >> 5;
  const int lane = threadIdx.x & 31;
  const int half = lane >> 4;
  const int l16  = lane & 15;

  const int Nt = (N + 15) >> 4;
  const int strips = (Nt + JT - 1) / JT;
  const int Mt = M >> 4;
  int sid = blockIdx.x * 8 + wave;  // 8 waves per 256-thread block
  if (sid >= Mt * strips) return;
  const int mt = sid / strips, st = sid % strips;
  const int m0 = mt << 4, n0 = st * (JT * 16);

  // Per-lane bases; all other addressing is compile-time immediates.
  const __bf16* abase = Ai + (long)(m0 + l16) * (2 * K) + half * 16;
  const __bf16* wbase = Wi + (long)min(n0 + l16, N - 1) * (2 * K) + half * 16;

  v8f acc0[JT];
  v8f acc1[JT];
#pragma unroll
  for (int j = 0; j < JT; ++j) { acc0[j] = (v8f){}; acc1[j] = (v8f){}; }

  for (int k0 = 0; k0 < K; k0 += 32, abase += 32, wbase += 32) {
    v16bf a_hi, a_lo;
    ((v8bf*)&a_hi)[0] = *(const v8bf*)(abase);
    ((v8bf*)&a_hi)[1] = *(const v8bf*)(abase + 8);
    ((v8bf*)&a_lo)[0] = *(const v8bf*)(abase + K);
    ((v8bf*)&a_lo)[1] = *(const v8bf*)(abase + K + 8);
#pragma unroll
    for (int j = 0; j < JT; ++j) {
      const __bf16* wr = wbase + (long)j * 16 * (2 * K);  // literal offset
      v16bf b_hi = *(const v16bf*)(wr);
      v16bf b_lo = *(const v16bf*)(wr + K);
      acc0[j] = __builtin_amdgcn_wmma_f32_16x16x32_bf16(
          false, a_hi, false, b_hi, (short)0, acc0[j], false, false);
      acc1[j] = __builtin_amdgcn_wmma_f32_16x16x32_bf16(
          false, a_hi, false, b_lo, (short)0, acc1[j], false, false);
      acc1[j] = __builtin_amdgcn_wmma_f32_16x16x32_bf16(
          false, a_lo, false, b_hi, (short)0, acc1[j], false, false);
    }
  }

#pragma unroll
  for (int j = 0; j < JT; ++j) {
    int n = n0 + j * 16 + l16;
    if (n < N) {
      float bv = bias[n];
#pragma unroll
      for (int r = 0; r < 8; ++r) {
        int m = m0 + r + half * 8;
        float v = acc0[j][r] + acc1[j][r] + bv;
        if (ACT) v = fmaxf(v, 0.0f);
        out[(long)m * N + n] = v;
      }
    }
  }
}

// ---------------------------------------------------------------------------
extern "C" void kernel_launch(void* const* d_in, const int* in_sizes, int n_in,
                              void* d_out, int out_size, void* d_ws, size_t ws_size,
                              hipStream_t stream) {
  (void)in_sizes; (void)n_in; (void)out_size; (void)ws_size;
  const float* input     = (const float*)d_in[0];
  const float* rois      = (const float*)d_in[1];
  const float* rescale_w = (const float*)d_in[2];
  const float* rescale_b = (const float*)d_in[3];
  const float* off_w1    = (const float*)d_in[4];
  const float* off_b1    = (const float*)d_in[5];
  const float* off_w2    = (const float*)d_in[6];
  const float* off_b2    = (const float*)d_in[7];
  const float* off_w3    = (const float*)d_in[8];
  const float* off_b3    = (const float*)d_in[9];

  float* out   = (float*)d_out;
  float* x_cls = out;                        // (512, 256, 7, 7)
  float* x_reg = out + (long)NROIS * FDIM;   // (512, 256, 7, 7)

  // Workspace carve-up (256B-aligned regions), ~152 MB total.
  char* ws = (char*)d_ws;
  size_t o = 0;
  auto carve = [&](size_t bytes) {
    char* p = ws + o;
    o = (o + bytes + 255) & ~(size_t)255;
    return p;
  };
  const long W1E = (long)DFC * FDIM;     // 12,845,056
  const long W2E = (long)DFC * DFC;      //  1,048,576
  const long W3E = (long)RESC * DFC;     //    100,352
  const long AE  = (long)NROIS * FDIM;   //  6,422,528
  const long HE  = (long)NROIS * DFC;    //    524,288

  float*  feat_nhwc = (float*)carve((size_t)NB * NH * NW * CCH * 4);  // 62.2 MB
  float*  new_rois  = (float*)carve((size_t)NROIS * 5 * 4);
  float*  h1        = (float*)carve(HE * 4);
  float*  h2        = (float*)carve(HE * 4);
  float*  off       = (float*)carve((size_t)NROIS * RESC * 4);
  __bf16* w1i = (__bf16*)carve(2 * W1E * 2);   // hi|lo row-interleaved
  __bf16* w2i = (__bf16*)carve(2 * W2E * 2);
  __bf16* w3i = (__bf16*)carve(2 * W3E * 2);
  __bf16* fai = (__bf16*)carve(2 * AE * 2);
  __bf16* h1i = (__bf16*)carve(2 * HE * 2);
  __bf16* h2i = (__bf16*)carve(2 * HE * 2);

  // 1) NCHW -> NHWC; weight splits (no dependency on the pools).
  nchw_to_nhwc_kernel<<<2048, 256, 0, stream>>>(input, feat_nhwc);
  split_bf16_rows_kernel<FDIM><<<2048, 256, 0, stream>>>(off_w1, w1i, W1E / 8);
  split_bf16_rows_kernel<DFC><<<1024, 256, 0, stream>>>(off_w2, w2i, W2E / 8);
  split_bf16_rows_kernel<DFC><<<128,  256, 0, stream>>>(off_w3, w3i, W3E / 8);

  // 2) x_cls = deform_roi_pool(input, rois, None)
  deform_roi_pool_kernel<<<NROIS * OH * OW, CCH, 0, stream>>>(
      feat_nhwc, rois, nullptr, x_cls);

  // 3) new_rois from rescale[:,0:4] (only 4 of 98 outputs are used).
  rescale_rois_kernel<<<NROIS, 256, 0, stream>>>(
      x_cls, rescale_w, rescale_b, rois, new_rois);

  // 4) h1 = relu(flat @ W1^T + b1)   [512 x 12544] x [12544 x 1024]
  split_bf16_rows_kernel<FDIM><<<2048, 256, 0, stream>>>(x_cls, fai, AE / 8);
  {
    int strips = (DFC / 16) / 4;                       // 16
    int waves  = (NROIS / 16) * strips;                // 512
    gemm_nt_bf16x3_kernel<1, 4, FDIM><<<(waves + 7) / 8, 256, 0, stream>>>(
        fai, w1i, off_b1, h1, NROIS, DFC);
  }
  // 5) h2 = relu(h1 @ W2^T + b2)     [512 x 1024] x [1024 x 1024]
  split_bf16_rows_kernel<DFC><<<256, 256, 0, stream>>>(h1, h1i, HE / 8);
  {
    int strips = (DFC / 16) / 4;
    int waves  = (NROIS / 16) * strips;
    gemm_nt_bf16x3_kernel<1, 4, DFC><<<(waves + 7) / 8, 256, 0, stream>>>(
        h1i, w2i, off_b2, h2, NROIS, DFC);
  }
  // 6) off = h2 @ W3^T + b3          [512 x 1024] x [1024 x 98]
  split_bf16_rows_kernel<DFC><<<256, 256, 0, stream>>>(h2, h2i, HE / 8);
  {
    int strips = (RESC + 15) / 16;                     // 7 (JT=1)
    int waves  = (NROIS / 16) * strips;                // 224
    gemm_nt_bf16x3_kernel<0, 1, DFC><<<(waves + 7) / 8, 256, 0, stream>>>(
        h2i, w3i, off_b3, off, NROIS, RESC);
  }

  // 7) x_reg = deform_roi_pool(input, new_rois, off)
  deform_roi_pool_kernel<<<NROIS * OH * OW, CCH, 0, stream>>>(
      feat_nhwc, new_rois, off, x_reg);
}
